// FakeNewsNet_31129922961619
// MI455X (gfx1250) — compile-verified, matrix operand order checked by reading, not run on hardware
//
#include <hip/hip_runtime.h>
#include <math.h>

#define NN   50000
#define EE   800000
#define ETOT (EE + NN)
#define BB   128
#define INDIM 310
#define HD   128

typedef __attribute__((ext_vector_type(16))) _Float16 v16h;
typedef __attribute__((ext_vector_type(8)))  float    v8f;

// ---------------------------------------------------------------- utilities
__global__ void fill_kernel(float* __restrict__ p, float v, int n) {
    int i = blockIdx.x * blockDim.x + threadIdx.x;
    if (i < n) p[i] = v;
}

__device__ __forceinline__ float lrelu(float v) { return v > 0.f ? v : 0.2f * v; }

__device__ __forceinline__ void atomicMaxFloat(float* addr, float val) {
    // monotone bit trick; works with init = -1e30
    if (val >= 0.f) atomicMax((int*)addr, __float_as_int(val));
    else            atomicMin((unsigned int*)addr, (unsigned int)__float_as_int(val));
}

// ---------------------------------------------------------------- WMMA GEMM
// C[M,128] = A[M,K] @ W[K,128], f32 in/out, f16 WMMA, f32 accumulate.
// REQUIRES M % 16 == 0 (all call sites: 50000, 128). Optional rowidx gathers
// rows of A. Block = 256 threads = 8 waves; each wave owns a 16-column slice.
// LDS tiles are in *fragment order* so each lane reads its 32-byte fragment
// with two ds_load_b128. Software pipeline: global loads for chunk k+1 are
// issued before the WMMA of chunk k and only consumed (cvt + ds_store) after
// it, hiding vmem latency behind the matrix op; one barrier per K-step.
//
// Fragment map (ISA 7.12.2, 16-bit A 16x32): element K (0..31) of row m sits
// at lane = m + 16*((K>>3)&1), halfword e = ((K>>4)<<3) + (K&7).
__global__ void gemm_wmma_f16(const float* __restrict__ A,
                              const float* __restrict__ W,
                              float* __restrict__ C,
                              int M, int K,
                              const int* __restrict__ rowidx) {
    // stride 40 halfs = 80 B: 16-byte aligned rows, bank-decorrelated
    __shared__ __align__(16) _Float16 Af[2][32][40];      //  5 KB
    __shared__ __align__(16) _Float16 Wf[2][8][32][40];   // 40 KB

    const int tid  = threadIdx.x;
    const int lane = tid & 31;
    const int wave = tid >> 5;
    const int row0 = blockIdx.x * 16;
    const int col0 = wave * 16;

    // ---- A staging: thread owns k=kA for rows mA0 and mA0+8 (loop-invariant)
    const int kA  = tid & 31;
    const int mA0 = tid >> 5;          // 0..7
    const int mA1 = mA0 + 8;           // 8..15
    const int hA  = (kA >> 3) & 1;
    const int eA  = ((kA >> 4) << 3) + (kA & 7);
    const int lA0 = mA0 + 16 * hA;
    const int lA1 = mA1 + 16 * hA;
    const int grow0 = row0 + mA0, grow1 = row0 + mA1;
    const float* pa0 = A + (size_t)(rowidx ? rowidx[grow0] : grow0) * K;
    const float* pa1 = A + (size_t)(rowidx ? rowidx[grow1] : grow1) * K;

    // ---- W staging: thread owns 16 consecutive cols of one K-row per chunk
    const int wW = tid & 7;            // destination wave slice
    const int rW = tid >> 3;           // K row within chunk, 0..31
    const int hW = (rW >> 3) & 1;
    const int eW = ((rW >> 4) << 3) + (rW & 7);
    const int c0 = wW * 16;

    // staged-chunk registers (held across the WMMA to hide vmem latency)
    float  rA0v, rA1v;
    float4 rw0, rw1, rw2, rw3;

    auto load_chunk = [&](int kc) {      // issue global loads, branchless
        int k  = kc + kA;  if (k  > K - 1) k  = K - 1;   // clamp, in-bounds
        rA0v = pa0[k];
        rA1v = pa1[k];
        int kg = kc + rW;  if (kg > K - 1) kg = K - 1;
        const float4* p = (const float4*)(W + (size_t)kg * HD + c0);
        rw0 = p[0]; rw1 = p[1]; rw2 = p[2]; rw3 = p[3];
        if ((kc + 64) < K) __builtin_prefetch(pa0 + kc + 64, 0, 1);
    };

    auto store_chunk = [&](int kc, int buf) {   // cvt + ds_store (waits here)
        const bool okA = (kc + kA) < K;
        Af[buf][lA0][eA] = (_Float16)(okA ? rA0v : 0.f);
        Af[buf][lA1][eA] = (_Float16)(okA ? rA1v : 0.f);
        const bool okW = (kc + rW) < K;
        float vals[16] = { rw0.x, rw0.y, rw0.z, rw0.w, rw1.x, rw1.y, rw1.z, rw1.w,
                           rw2.x, rw2.y, rw2.z, rw2.w, rw3.x, rw3.y, rw3.z, rw3.w };
        #pragma unroll
        for (int n = 0; n < 16; ++n)
            Wf[buf][wW][n + (hW << 4)][eW] = (_Float16)(okW ? vals[n] : 0.f);
    };

    v8f c = {};
    load_chunk(0);
    store_chunk(0, 0);
    __syncthreads();
    int cur = 0;
    for (int kc = 0; kc < K; kc += 32) {
        const bool more = (kc + 32) < K;
        if (more) load_chunk(kc + 32);          // vmem issued, no wait
        v16h a = *(const v16h*)(&Af[cur][lane][0]);         // 2x ds_load_b128
        v16h b = *(const v16h*)(&Wf[cur][wave][lane][0]);   // 2x ds_load_b128
        c = __builtin_amdgcn_wmma_f32_16x16x32_f16(
                false, a, false, b, (short)0, c, false, false);
        if (more) store_chunk(kc + 32, cur ^ 1);   // wait + cvt + ds_store
        __syncthreads();
        cur ^= 1;
    }

    const int hlf = lane >> 4;
    const int nc  = lane & 15;
    #pragma unroll
    for (int r = 0; r < 8; ++r)
        C[(size_t)(row0 + r + 8 * hlf) * HD + col0 + nc] = c[r];
}

// ---------------------------------------------------------------- attention
// alpha_src[n,h] = sum_d xw[n,h,d]*a_src[h,d]   (H=2, D=64), wave per node
__global__ void alpha_kernel(const float* __restrict__ xw,
                             const float* __restrict__ a_src,
                             const float* __restrict__ a_dst,
                             float* __restrict__ asrc,
                             float* __restrict__ adst) {
    int lane = threadIdx.x & 31;
    int node = blockIdx.x * 8 + (threadIdx.x >> 5);
    if (node >= NN) return;
    const float* row = xw + (size_t)node * HD;
    float x0 = row[lane], x1 = row[32 + lane], x2 = row[64 + lane], x3 = row[96 + lane];
    float s0 = x0 * a_src[lane] + x1 * a_src[32 + lane];
    float s1 = x2 * a_src[64 + lane] + x3 * a_src[96 + lane];
    float d0 = x0 * a_dst[lane] + x1 * a_dst[32 + lane];
    float d1 = x2 * a_dst[64 + lane] + x3 * a_dst[96 + lane];
    #pragma unroll
    for (int off = 16; off > 0; off >>= 1) {
        s0 += __shfl_down(s0, off);
        s1 += __shfl_down(s1, off);
        d0 += __shfl_down(d0, off);
        d1 += __shfl_down(d1, off);
    }
    if (lane == 0) {
        asrc[node * 2]     = s0;
        asrc[node * 2 + 1] = s1;
        adst[node * 2]     = d0;
        adst[node * 2 + 1] = d1;
    }
}

__device__ __forceinline__ void edge_sd(const int* __restrict__ ei, int i, int& s, int& d) {
    if (i < EE) { s = ei[i]; d = ei[EE + i]; }
    else        { s = i - EE; d = i - EE; }   // self loops
}

__global__ void edge_max_kernel(const int* __restrict__ ei,
                                const float* __restrict__ asrc,
                                const float* __restrict__ adst,
                                float* __restrict__ mmax) {
    int i = blockIdx.x * blockDim.x + threadIdx.x;
    if (i >= ETOT) return;
    int s, d; edge_sd(ei, i, s, d);
    #pragma unroll
    for (int h = 0; h < 2; ++h) {
        float e = lrelu(asrc[s * 2 + h] + adst[d * 2 + h]);
        atomicMaxFloat(&mmax[d * 2 + h], e);
    }
}

__global__ void edge_den_kernel(const int* __restrict__ ei,
                                const float* __restrict__ asrc,
                                const float* __restrict__ adst,
                                const float* __restrict__ mmax,
                                float* __restrict__ den) {
    int i = blockIdx.x * blockDim.x + threadIdx.x;
    if (i >= ETOT) return;
    int s, d; edge_sd(ei, i, s, d);
    #pragma unroll
    for (int h = 0; h < 2; ++h) {
        float e = lrelu(asrc[s * 2 + h] + adst[d * 2 + h]);
        atomicAdd(&den[d * 2 + h], expf(e - mmax[d * 2 + h]));
    }
}

// wave per edge: scatter xw[src]*alpha into acc[dst], 4 features/lane
__global__ void edge_scatter_kernel(const int* __restrict__ ei,
                                    const float* __restrict__ asrc,
                                    const float* __restrict__ adst,
                                    const float* __restrict__ mmax,
                                    const float* __restrict__ den,
                                    const float* __restrict__ xw,
                                    float* __restrict__ acc) {
    int lane = threadIdx.x & 31;
    int e = blockIdx.x * 8 + (threadIdx.x >> 5);
    if (e >= ETOT) return;
    int s, d; edge_sd(ei, e, s, d);
    float al0 = expf(lrelu(asrc[s * 2]     + adst[d * 2])     - mmax[d * 2])
                / (den[d * 2] + 1e-16f);
    float al1 = expf(lrelu(asrc[s * 2 + 1] + adst[d * 2 + 1]) - mmax[d * 2 + 1])
                / (den[d * 2 + 1] + 1e-16f);
    const float* srow = xw + (size_t)s * HD;
    float* drow = acc + (size_t)d * HD;
    #pragma unroll
    for (int j = 0; j < 4; ++j) {
        int c = lane + 32 * j;
        atomicAdd(&drow[c], srow[c] * (c < 64 ? al0 : al1));
    }
}

__global__ void bias_relu_kernel(float* __restrict__ buf,
                                 const float* __restrict__ bias, int count4) {
    int i = blockIdx.x * blockDim.x + threadIdx.x;   // index in float4 units
    if (i >= count4) return;
    float4 v = ((const float4*)buf)[i];
    const float4 b = ((const float4*)bias)[i & (HD / 4 - 1)];
    v.x = fmaxf(v.x + b.x, 0.f);
    v.y = fmaxf(v.y + b.y, 0.f);
    v.z = fmaxf(v.z + b.z, 0.f);
    v.w = fmaxf(v.w + b.w, 0.f);
    ((float4*)buf)[i] = v;
}

// ---------------------------------------------------------------- pooling
__global__ void pool_kernel(const float* __restrict__ h,
                            const int* __restrict__ batch,
                            float* __restrict__ pooled,
                            float* __restrict__ cnt) {
    int lane = threadIdx.x & 31;
    int node = blockIdx.x * 8 + (threadIdx.x >> 5);
    if (node >= NN) return;
    int g = batch[node];
    const float* row = h + (size_t)node * HD;
    float* prow = pooled + (size_t)g * HD;
    #pragma unroll
    for (int j = 0; j < 4; ++j) {
        int c = lane + 32 * j;
        atomicAdd(&prow[c], row[c]);
    }
    if (lane == 0) atomicAdd(&cnt[g], 1.0f);
}

__global__ void divide_kernel(float* __restrict__ pooled,
                              const float* __restrict__ cnt) {
    int i = blockIdx.x * blockDim.x + threadIdx.x;
    if (i >= BB * HD) return;
    pooled[i] /= cnt[i >> 7];
}

__global__ void root_kernel(const int* __restrict__ batch, int* __restrict__ root) {
    int n = blockIdx.x * blockDim.x + threadIdx.x;
    if (n >= NN) return;
    if (n == 0 || batch[n] != batch[n - 1]) root[batch[n]] = n;
}

__global__ void final_kernel(const float* __restrict__ g,
                             const float* __restrict__ news,
                             const float* __restrict__ Wc,
                             const float* __restrict__ bc,
                             float* __restrict__ out) {
    int b = threadIdx.x;
    if (b >= BB) return;
    float acc = bc[0];
    #pragma unroll 4
    for (int j = 0; j < HD; ++j)
        acc += g[b * HD + j] * Wc[j] + news[b * HD + j] * Wc[HD + j];
    out[b] = 1.f / (1.f + expf(-acc));
}

// ---------------------------------------------------------------- driver
extern "C" void kernel_launch(void* const* d_in, const int* in_sizes, int n_in,
                              void* d_out, int out_size, void* d_ws, size_t ws_size,
                              hipStream_t stream) {
    const float* x    = (const float*)d_in[0];
    const float* W0   = (const float*)d_in[1];
    const float* as0  = (const float*)d_in[2];
    const float* ad0  = (const float*)d_in[3];
    const float* b0   = (const float*)d_in[4];
    const float* W1   = (const float*)d_in[5];
    const float* as1  = (const float*)d_in[6];
    const float* ad1  = (const float*)d_in[7];
    const float* b1   = (const float*)d_in[8];
    const float* Wn   = (const float*)d_in[9];
    const float* bn   = (const float*)d_in[10];
    const float* Wr   = (const float*)d_in[11];
    const float* br   = (const float*)d_in[12];
    const float* Wc   = (const float*)d_in[13];
    const float* bc   = (const float*)d_in[14];
    const int*   ei   = (const int*)d_in[15];
    const int*   batch= (const int*)d_in[16];
    float* out = (float*)d_out;

    char* ws = (char*)d_ws;
    float* bufA  = (float*)ws;  ws += (size_t)NN * HD * 4;   // xw
    float* bufB  = (float*)ws;  ws += (size_t)NN * HD * 4;   // acc / h
    float* asrc  = (float*)ws;  ws += (size_t)NN * 2 * 4;
    float* adst  = (float*)ws;  ws += (size_t)NN * 2 * 4;
    float* mmax  = (float*)ws;  ws += (size_t)NN * 2 * 4;
    float* den   = (float*)ws;  ws += (size_t)NN * 2 * 4;
    float* pooled= (float*)ws;  ws += (size_t)BB * HD * 4;
    float* cnt   = (float*)ws;  ws += (size_t)BB * 4;
    float* gbuf  = (float*)ws;  ws += (size_t)BB * HD * 4;
    float* nbuf  = (float*)ws;  ws += (size_t)BB * HD * 4;
    int*   root  = (int*)ws;    ws += (size_t)BB * 4;

    const dim3 blk(256);
    const int gemm_grid_N = NN / 16;       // 3125 (exact)
    const int edge_grid_t = (ETOT + 255) / 256;
    const int edge_grid_w = (ETOT + 7) / 8;
    const int node_grid_w = (NN + 7) / 8;
    const int feat4_grid  = (NN * HD / 4 + 255) / 256;
    const int feat_grid   = (NN * HD + 255) / 256;
    const int n2_grid     = (NN * 2 + 255) / 256;

    // ---------------- GAT layer 0 ----------------
    gemm_wmma_f16<<<gemm_grid_N, blk, 0, stream>>>(x, W0, bufA, NN, INDIM, nullptr);
    alpha_kernel<<<node_grid_w, blk, 0, stream>>>(bufA, as0, ad0, asrc, adst);
    fill_kernel<<<n2_grid, blk, 0, stream>>>(mmax, -1e30f, NN * 2);
    fill_kernel<<<n2_grid, blk, 0, stream>>>(den, 0.f, NN * 2);
    fill_kernel<<<feat_grid, blk, 0, stream>>>(bufB, 0.f, NN * HD);
    edge_max_kernel<<<edge_grid_t, blk, 0, stream>>>(ei, asrc, adst, mmax);
    edge_den_kernel<<<edge_grid_t, blk, 0, stream>>>(ei, asrc, adst, mmax, den);
    edge_scatter_kernel<<<edge_grid_w, blk, 0, stream>>>(ei, asrc, adst, mmax, den, bufA, bufB);
    bias_relu_kernel<<<feat4_grid, blk, 0, stream>>>(bufB, b0, NN * HD / 4);  // h1 in bufB

    // ---------------- GAT layer 1 ----------------
    gemm_wmma_f16<<<gemm_grid_N, blk, 0, stream>>>(bufB, W1, bufA, NN, HD, nullptr);
    alpha_kernel<<<node_grid_w, blk, 0, stream>>>(bufA, as1, ad1, asrc, adst);
    fill_kernel<<<n2_grid, blk, 0, stream>>>(mmax, -1e30f, NN * 2);
    fill_kernel<<<n2_grid, blk, 0, stream>>>(den, 0.f, NN * 2);
    fill_kernel<<<feat_grid, blk, 0, stream>>>(bufB, 0.f, NN * HD);  // h1 no longer needed
    edge_max_kernel<<<edge_grid_t, blk, 0, stream>>>(ei, asrc, adst, mmax);
    edge_den_kernel<<<edge_grid_t, blk, 0, stream>>>(ei, asrc, adst, mmax, den);
    edge_scatter_kernel<<<edge_grid_w, blk, 0, stream>>>(ei, asrc, adst, mmax, den, bufA, bufB);
    bias_relu_kernel<<<feat4_grid, blk, 0, stream>>>(bufB, b1, NN * HD / 4);  // h2 in bufB

    // ---------------- pooling + head ----------------
    fill_kernel<<<(BB * HD + 255) / 256, blk, 0, stream>>>(pooled, 0.f, BB * HD);
    fill_kernel<<<1, blk, 0, stream>>>(cnt, 0.f, BB);
    pool_kernel<<<node_grid_w, blk, 0, stream>>>(bufB, batch, pooled, cnt);
    divide_kernel<<<(BB * HD + 255) / 256, blk, 0, stream>>>(pooled, cnt);
    root_kernel<<<(NN + 255) / 256, blk, 0, stream>>>(batch, root);

    gemm_wmma_f16<<<BB / 16, blk, 0, stream>>>(pooled, Wr, gbuf, BB, HD, nullptr);
    bias_relu_kernel<<<(BB * HD / 4 + 255) / 256, blk, 0, stream>>>(gbuf, br, BB * HD / 4);

    gemm_wmma_f16<<<BB / 16, blk, 0, stream>>>(x, Wn, nbuf, BB, INDIM, root);
    bias_relu_kernel<<<(BB * HD / 4 + 255) / 256, blk, 0, stream>>>(nbuf, bn, BB * HD / 4);

    final_kernel<<<1, 128, 0, stream>>>(gbuf, nbuf, Wc, bc, out);
}
